// TransformerBlock_17480516894759
// MI455X (gfx1250) — compile-verified, matrix operand order checked by reading, not run on hardware
//
#include <hip/hip_runtime.h>
#include <hip/hip_bf16.h>
#include <stdint.h>

// ---------------------------------------------------------------------------
// Problem constants (match reference)
// ---------------------------------------------------------------------------
#define BATCH  2
#define SEQ    2048
#define DMODEL 1024
#define NH     16
#define DHEAD  64
#define FFDIM  4096
#define EPSLN  1e-6f

// ---------------------------------------------------------------------------
// WMMA types (CDNA5 / gfx1250, wave32)
// ---------------------------------------------------------------------------
typedef __attribute__((ext_vector_type(16))) __bf16 v16bf;
typedef __attribute__((ext_vector_type(8)))  float  v8f;

union Frag {
    v16bf v;
    uint4 q[2];
};

__device__ __forceinline__ __bf16 f2bf(float f) {
    union { float f; uint32_t u; } c; c.f = f;
    uint32_t r = (c.u + 0x7FFFu + ((c.u >> 16) & 1u)) >> 16;   // RNE
    union { uint16_t s; __bf16 b; } o; o.s = (uint16_t)r;
    return o.b;
}

// ---------------------------------------------------------------------------
// Generic bf16-WMMA GEMM: C[M,N] = A[M,K] * B[K,N]  (A,B fp32 in memory,
// converted to bf16 in LDS; fp32 accumulate via v_wmma_f32_16x16x32_bf16)
//
// Block tile: 128x128, K-step 32, 256 threads (8 wave32), wave tile 64x32.
// LDS rows padded to BK+8 bf16 (80B) so per-lane fragment reads are two
// 16B-aligned ds_load_b128, matching the 16-bit A/B operand layout:
//   lane l: row/col = l&15, half = l>>4; chunk0 = K[half*8 .. +7],
//   chunk1 = K[16+half*8 .. +7]  (per K-step of 32).
// ---------------------------------------------------------------------------
#define BM 128
#define BN 128
#define BK 32
#define GT 256
#define LDAP 40   // BK + 8 pad -> 80B row stride, 16B aligned rows

enum { EPI_F32 = 0, EPI_BIAS = 1, EPI_BIAS_RELU = 2, EPI_QKPACK = 3, EPI_VTPACK = 4 };

template<int EPI, bool TRANSB>
__launch_bounds__(GT)
__global__ void gemm_bf16_wmma(const float* __restrict__ A,
                               const float* __restrict__ B,
                               const float* __restrict__ bias,
                               float* __restrict__ C,
                               __bf16* __restrict__ Cb,
                               int M, int N, int K, float scale)
{
    __shared__ __bf16 As[BM][LDAP];
    __shared__ __bf16 Bs[BN][LDAP];

    const int tid  = threadIdx.x;
    const int m0   = blockIdx.y * BM;
    const int n0   = blockIdx.x * BN;
    const int wid  = tid >> 5;
    const int lane = tid & 31;
    const int hf   = lane >> 4;
    const int ln   = lane & 15;
    const int wm   = (wid >> 2) * 64;   // 2 wave rows
    const int wn   = (wid & 3) * 32;    // 4 wave cols

    v8f acc[4][2] = {};

    for (int kt = 0; kt < K; kt += BK) {
        // ---- stage A tile (128 x 32 fp32 -> bf16) --------------------------
        #pragma unroll
        for (int i = 0; i < 4; ++i) {
            int idx = i * GT + tid;              // 0..1023 float4 slots
            int r = idx >> 3, c = (idx & 7) << 2;
            const float4 a4 = *(const float4*)(A + (size_t)(m0 + r) * K + kt + c);
            As[r][c + 0] = f2bf(a4.x); As[r][c + 1] = f2bf(a4.y);
            As[r][c + 2] = f2bf(a4.z); As[r][c + 3] = f2bf(a4.w);
        }
        // ---- stage B tile ---------------------------------------------------
        if (TRANSB) {
            // B given as [N,K] row-major (Wo): store Bs[n][k] contiguously
            #pragma unroll
            for (int i = 0; i < 4; ++i) {
                int idx = i * GT + tid;
                int r = idx >> 3, c = (idx & 7) << 2;
                const float4 b4 = *(const float4*)(B + (size_t)(n0 + r) * K + kt + c);
                Bs[r][c + 0] = f2bf(b4.x); Bs[r][c + 1] = f2bf(b4.y);
                Bs[r][c + 2] = f2bf(b4.z); Bs[r][c + 3] = f2bf(b4.w);
            }
        } else {
            // B [K,N] row-major: transpose while staging into Bs[n][k]
            #pragma unroll
            for (int i = 0; i < 4; ++i) {
                int idx = i * GT + tid;
                int k = idx >> 5, c = (idx & 31) << 2;
                const float4 b4 = *(const float4*)(B + (size_t)(kt + k) * N + n0 + c);
                Bs[c + 0][k] = f2bf(b4.x); Bs[c + 1][k] = f2bf(b4.y);
                Bs[c + 2][k] = f2bf(b4.z); Bs[c + 3][k] = f2bf(b4.w);
            }
        }
        if (kt + BK < K) {  // hint next A tile into cache (global_prefetch)
            __builtin_prefetch(A + (size_t)(m0 + (tid >> 3)) * K + kt + BK, 0, 1);
        }
        __syncthreads();

        // ---- fragments + 8 WMMAs -------------------------------------------
        Frag af[4], bfr[2];
        #pragma unroll
        for (int mt = 0; mt < 4; ++mt) {
            const __bf16* p = &As[wm + mt * 16 + ln][0];
            af[mt].q[0] = *(const uint4*)(p + hf * 8);
            af[mt].q[1] = *(const uint4*)(p + 16 + hf * 8);
        }
        #pragma unroll
        for (int nt = 0; nt < 2; ++nt) {
            const __bf16* p = &Bs[wn + nt * 16 + ln][0];
            bfr[nt].q[0] = *(const uint4*)(p + hf * 8);
            bfr[nt].q[1] = *(const uint4*)(p + 16 + hf * 8);
        }
        #pragma unroll
        for (int mt = 0; mt < 4; ++mt)
            #pragma unroll
            for (int nt = 0; nt < 2; ++nt)
                acc[mt][nt] = __builtin_amdgcn_wmma_f32_16x16x32_bf16(
                    false, af[mt].v, false, bfr[nt].v, (short)0, acc[mt][nt], false, false);
        __syncthreads();
    }

    // ---- epilogue: C/D layout -> element (m,n): vgpr r = m&7, lane = n + 16*(m>>3)
    #pragma unroll
    for (int mt = 0; mt < 4; ++mt)
    #pragma unroll
    for (int nt = 0; nt < 2; ++nt)
    #pragma unroll
    for (int r = 0; r < 8; ++r) {
        int row = m0 + wm + mt * 16 + r + hf * 8;
        int col = n0 + wn + nt * 16 + ln;
        float v = acc[mt][nt][r];
        if (EPI == EPI_BIAS || EPI == EPI_BIAS_RELU) v += bias[col];
        if (EPI == EPI_BIAS_RELU) v = fmaxf(v, 0.0f);
        if (EPI == EPI_F32 || EPI == EPI_BIAS || EPI == EPI_BIAS_RELU) {
            C[(size_t)row * N + col] = v;
        } else if (EPI == EPI_QKPACK) {
            // [B*T, H*64] fp32 -> [B,H,T,64] bf16 (Q pre-scaled by 1/sqrt(DK))
            int b = row >> 11, t = row & (SEQ - 1);
            int h = col >> 6,  dd = col & 63;
            Cb[(((size_t)(b * NH + h) * SEQ + t) << 6) + dd] = f2bf(v * scale);
        } else { // EPI_VTPACK: -> [B,H,64,T] bf16 (V transposed)
            int b = row >> 11, t = row & (SEQ - 1);
            int h = col >> 6,  dd = col & 63;
            Cb[((size_t)(b * NH + h) * 64 + dd) * SEQ + t] = f2bf(v);
        }
    }
}

// ---------------------------------------------------------------------------
// Causal flash attention: grid (T/64, B*H), 128 threads (4 waves).
// Wave w owns q rows qtile*64 + w*16 .. +15; streams K/V in 64-wide chunks
// (only sc <= qtile: causal), online softmax, ctx accumulated in fp32.
// P (C-layout) round-trips through per-wave LDS tile to become an A operand.
// ---------------------------------------------------------------------------
#define AT 128
#define KPAD 72   // 64 + 8 pad -> 144B row stride, 16B aligned rows

__launch_bounds__(AT)
__global__ void flash_attn_kernel(const __bf16* __restrict__ qg,
                                  const __bf16* __restrict__ kg,
                                  const __bf16* __restrict__ vtg,
                                  float* __restrict__ ctx)
{
    __shared__ __bf16 Ks[64][KPAD];       // K chunk   [s][dk]
    __shared__ __bf16 Vs[64][KPAD];       // Vt chunk  [dv][s]
    __shared__ __bf16 Ps[4][16][KPAD];    // per-wave P [t][s]

    const int tid   = threadIdx.x;
    const int wid   = tid >> 5;
    const int lane  = tid & 31;
    const int hf    = lane >> 4;
    const int ln    = lane & 15;
    const int bh    = blockIdx.y;         // b*NH + h
    const int qtile = blockIdx.x;
    const int qrow0 = qtile * 64 + wid * 16;

    // Q fragments (A operand, K = DHEAD = 64 -> two K-steps of 32)
    const __bf16* qrow = qg + ((size_t)bh * SEQ + qrow0 + ln) * DHEAD;
    Frag aq[2];
    #pragma unroll
    for (int ks = 0; ks < 2; ++ks) {
        aq[ks].q[0] = *(const uint4*)(qrow + ks * 32 + hf * 8);
        aq[ks].q[1] = *(const uint4*)(qrow + ks * 32 + 16 + hf * 8);
    }

    float rowmax[8], rowsum[8];
    #pragma unroll
    for (int r = 0; r < 8; ++r) { rowmax[r] = -3.0e38f; rowsum[r] = 0.0f; }
    v8f cacc[4] = {};

    const __bf16* kbase = kg  + (size_t)bh * SEQ * DHEAD;
    const __bf16* vbase = vtg + (size_t)bh * DHEAD * SEQ;

    for (int sc = 0; sc <= qtile; ++sc) {
        const int s0 = sc * 64;
        // ---- stage K chunk [64][64] and Vt chunk [64][64] ------------------
        #pragma unroll
        for (int i = 0; i < 8; ++i) {
            int idx = i * AT + tid;           // 0..1023 (8B units)
            int r = idx >> 4, c = (idx & 15) << 2;
            *(uint2*)&Ks[r][c] = *(const uint2*)(kbase + (size_t)(s0 + r) * DHEAD + c);
            *(uint2*)&Vs[r][c] = *(const uint2*)(vbase + (size_t)r * SEQ + s0 + c);
        }
        __syncthreads();

        // ---- S = Q * K^T  (16 x 64 per wave) -------------------------------
        v8f sacc[4] = {};
        #pragma unroll
        for (int nt = 0; nt < 4; ++nt) {
            #pragma unroll
            for (int ks = 0; ks < 2; ++ks) {
                Frag bk;
                const __bf16* p = &Ks[nt * 16 + ln][ks * 32];
                bk.q[0] = *(const uint4*)(p + hf * 8);
                bk.q[1] = *(const uint4*)(p + 16 + hf * 8);
                sacc[nt] = __builtin_amdgcn_wmma_f32_16x16x32_bf16(
                    false, aq[ks].v, false, bk.v, (short)0, sacc[nt], false, false);
            }
        }
        // ---- causal mask ----------------------------------------------------
        #pragma unroll
        for (int nt = 0; nt < 4; ++nt)
            #pragma unroll
            for (int r = 0; r < 8; ++r) {
                int s_idx = s0 + nt * 16 + ln;
                int t_idx = qrow0 + r + hf * 8;
                sacc[nt][r] = (s_idx > t_idx) ? -3.0e38f : sacc[nt][r];
            }
        // ---- online softmax (row stats live across 16 lanes of a half) -----
        #pragma unroll
        for (int r = 0; r < 8; ++r) {
            float mx = fmaxf(fmaxf(sacc[0][r], sacc[1][r]), fmaxf(sacc[2][r], sacc[3][r]));
            #pragma unroll
            for (int off = 1; off < 16; off <<= 1) mx = fmaxf(mx, __shfl_xor(mx, off, 32));
            float mnew = fmaxf(rowmax[r], mx);
            float corr = __expf(rowmax[r] - mnew);
            rowmax[r] = mnew;
            float ps = 0.0f;
            #pragma unroll
            for (int nt = 0; nt < 4; ++nt) {
                float p = __expf(sacc[nt][r] - mnew);
                ps += p;
                Ps[wid][r + hf * 8][nt * 16 + ln] = f2bf(p);
            }
            #pragma unroll
            for (int off = 1; off < 16; off <<= 1) ps += __shfl_xor(ps, off, 32);
            rowsum[r] = rowsum[r] * corr + ps;
            #pragma unroll
            for (int nt = 0; nt < 4; ++nt) cacc[nt][r] = cacc[nt][r] * corr;
        }
        // ---- ctx += P * V  (P reloaded in A layout from LDS) ---------------
        Frag ap[2];
        #pragma unroll
        for (int ks = 0; ks < 2; ++ks) {
            const __bf16* p = &Ps[wid][ln][ks * 32];
            ap[ks].q[0] = *(const uint4*)(p + hf * 8);
            ap[ks].q[1] = *(const uint4*)(p + 16 + hf * 8);
        }
        #pragma unroll
        for (int nt = 0; nt < 4; ++nt)
            #pragma unroll
            for (int ks = 0; ks < 2; ++ks) {
                Frag bv;
                const __bf16* p = &Vs[nt * 16 + ln][ks * 32];
                bv.q[0] = *(const uint4*)(p + hf * 8);
                bv.q[1] = *(const uint4*)(p + 16 + hf * 8);
                cacc[nt] = __builtin_amdgcn_wmma_f32_16x16x32_bf16(
                    false, ap[ks].v, false, bv.v, (short)0, cacc[nt], false, false);
            }
        __syncthreads();
    }

    // ---- normalize + store ctx as [B*T, H*DV] fp32 -------------------------
    const int b = bh / NH, h = bh % NH;
    #pragma unroll
    for (int r = 0; r < 8; ++r) {
        float inv = 1.0f / rowsum[r];
        int t = qrow0 + r + hf * 8;
        #pragma unroll
        for (int nt = 0; nt < 4; ++nt)
            ctx[(size_t)(b * SEQ + t) * DMODEL + h * 64 + nt * 16 + ln] = cacc[nt][r] * inv;
    }
}

// ---------------------------------------------------------------------------
// out = LayerNorm(a + b) * gamma + beta ; one 1024-wide row per 256-thr block
// ---------------------------------------------------------------------------
__launch_bounds__(256)
__global__ void add_ln_kernel(const float* __restrict__ a,
                              const float* __restrict__ b,
                              const float* __restrict__ gamma,
                              const float* __restrict__ beta,
                              float* __restrict__ out)
{
    __shared__ float red[16];
    const int row = blockIdx.x;
    const int tid = threadIdx.x;
    const size_t base = (size_t)row * DMODEL;

    float v[4]; float s = 0.0f, ss = 0.0f;
    #pragma unroll
    for (int j = 0; j < 4; ++j) {
        int idx = tid + j * 256;
        float x = a[base + idx] + b[base + idx];
        v[j] = x; s += x; ss += x * x;
    }
    #pragma unroll
    for (int off = 1; off < 32; off <<= 1) {
        s  += __shfl_xor(s,  off, 32);
        ss += __shfl_xor(ss, off, 32);
    }
    int wid = tid >> 5, lane = tid & 31;
    if (lane == 0) { red[wid] = s; red[8 + wid] = ss; }
    __syncthreads();
    if (wid == 0) {
        float s2  = (lane < 8) ? red[lane] : 0.0f;
        float ss2 = (lane < 8) ? red[8 + lane] : 0.0f;
        #pragma unroll
        for (int off = 1; off < 8; off <<= 1) {
            s2  += __shfl_xor(s2,  off, 32);
            ss2 += __shfl_xor(ss2, off, 32);
        }
        if (lane == 0) { red[0] = s2; red[8] = ss2; }
    }
    __syncthreads();
    float mu  = red[0] * (1.0f / DMODEL);
    float var = red[8] * (1.0f / DMODEL) - mu * mu;
    float rs  = rsqrtf(var + EPSLN);
    #pragma unroll
    for (int j = 0; j < 4; ++j) {
        int idx = tid + j * 256;
        out[base + idx] = (v[j] - mu) * rs * gamma[idx] + beta[idx];
    }
}

// ---------------------------------------------------------------------------
// Host-side orchestration
// ---------------------------------------------------------------------------
extern "C" void kernel_launch(void* const* d_in, const int* in_sizes, int n_in,
                              void* d_out, int out_size, void* d_ws, size_t ws_size,
                              hipStream_t stream)
{
    const float* x   = (const float*)d_in[0];
    const float* Wq  = (const float*)d_in[1];
    const float* Wk  = (const float*)d_in[2];
    const float* Wv  = (const float*)d_in[3];
    const float* Wo  = (const float*)d_in[4];
    const float* W1  = (const float*)d_in[5];
    const float* b1  = (const float*)d_in[6];
    const float* W2  = (const float*)d_in[7];
    const float* b2  = (const float*)d_in[8];
    const float* g1  = (const float*)d_in[9];
    const float* be1 = (const float*)d_in[10];
    const float* g2  = (const float*)d_in[11];
    const float* be2 = (const float*)d_in[12];
    float* out = (float*)d_out;

    const int M = BATCH * SEQ;   // 4096 token rows

    char* w = (char*)d_ws;
    auto take = [&](size_t bytes) -> char* {
        char* p = w; w += (bytes + 255) & ~(size_t)255; return p;
    };
    __bf16* qb  = (__bf16*)take((size_t)M * DMODEL * sizeof(__bf16)); //  8 MB
    __bf16* kb  = (__bf16*)take((size_t)M * DMODEL * sizeof(__bf16)); //  8 MB
    __bf16* vtb = (__bf16*)take((size_t)M * DMODEL * sizeof(__bf16)); //  8 MB
    float*  ctx = (float*)take((size_t)M * DMODEL * sizeof(float));   // 16 MB
    float*  ao  = (float*)take((size_t)M * DMODEL * sizeof(float));   // 16 MB
    float*  h   = (float*)take((size_t)M * DMODEL * sizeof(float));   // 16 MB
    float*  f1  = (float*)take((size_t)M * FFDIM  * sizeof(float));   // 64 MB
    float*  f2  = (float*)take((size_t)M * DMODEL * sizeof(float));   // 16 MB

    const dim3 blk(GT);
    const dim3 g_d(DMODEL / BN, M / BM);   // (8, 32)
    const dim3 g_ff(FFDIM / BN, M / BM);   // (32, 32)
    const float qscale = 0.125f;           // 1/sqrt(DK)

    // Q,K,V projections; pack straight to attention layouts in bf16
    gemm_bf16_wmma<EPI_QKPACK, false><<<g_d, blk, 0, stream>>>(x, Wq, nullptr, nullptr, qb,  M, DMODEL, DMODEL, qscale);
    gemm_bf16_wmma<EPI_QKPACK, false><<<g_d, blk, 0, stream>>>(x, Wk, nullptr, nullptr, kb,  M, DMODEL, DMODEL, 1.0f);
    gemm_bf16_wmma<EPI_VTPACK, false><<<g_d, blk, 0, stream>>>(x, Wv, nullptr, nullptr, vtb, M, DMODEL, DMODEL, 1.0f);

    // Causal flash attention
    flash_attn_kernel<<<dim3(SEQ / 64, BATCH * NH), dim3(AT), 0, stream>>>(qb, kb, vtb, ctx);

    // Output projection (Wo consumed transposed: [N,K] row-major)
    gemm_bf16_wmma<EPI_F32, true><<<g_d, blk, 0, stream>>>(ctx, Wo, nullptr, ao, nullptr, M, DMODEL, DMODEL, 1.0f);

    // h = LN(x + attn_out)
    add_ln_kernel<<<dim3(M), dim3(256), 0, stream>>>(x, ao, g1, be1, h);

    // FFN
    gemm_bf16_wmma<EPI_BIAS_RELU, false><<<g_ff, blk, 0, stream>>>(h,  W1, b1, f1, nullptr, M, FFDIM,  DMODEL, 1.0f);
    gemm_bf16_wmma<EPI_BIAS,      false><<<g_d,  blk, 0, stream>>>(f1, W2, b2, f2, nullptr, M, DMODEL, FFDIM,  1.0f);

    // out = LN(h + ffn)
    add_ln_kernel<<<dim3(M), dim3(256), 0, stream>>>(h, f2, g2, be2, out);
}
